// BiMambaEncoderLayer_V1_56710748176553
// MI455X (gfx1250) — matrix operand order repaired and checked by reading, not compiled, not run on hardware
//
#include <hip/hip_runtime.h>
#include <math.h>

// ---------------------------------------------------------------------------
// BiMamba encoder layer for MI455X (gfx1250, wave32).
// GEMMs run as bf16 WMMA (v_wmma_f32_16x16x32_bf16) with fp32 accumulate.
// 128x128 block tile, 8 waves, each wave 32x64 = 8 WMMAs per K-step.
// Global->LDS staging uses CDNA5 async loads (GLOBAL_LOAD_ASYNC_TO_LDS_B128,
// ASYNCcnt) when the builtins exist; falls back to b128 loads + ds_store_b128.
// ---------------------------------------------------------------------------

typedef __attribute__((ext_vector_type(16))) __bf16 v16bf;
typedef __attribute__((ext_vector_type(8)))  float  v8f;

union Frag16 { v16bf v; unsigned int u[8]; };

#define NBATCH 2
#define LSEQ   1024
#define NTOK   2048          // NBATCH * LSEQ
#define DMODEL 768
#define DINNER 1536
#define DSTATE 16
#define DTRANK 48
#define DBCW   80            // DTRANK + 2*DSTATE
#define DFF    3072

#if __has_builtin(__builtin_amdgcn_global_load_async_to_lds_b128) && \
    __has_builtin(__builtin_amdgcn_s_wait_asynccnt)
#define USE_ASYNC_LDS 1
// Builtin signature (from hipcc diagnostic): params are pointers to
// 'int __attribute__((vector_size(16)))' in AS(1) (global) and AS(3) (LDS).
typedef int v4i_vs __attribute__((__vector_size__(16)));
typedef __attribute__((address_space(1))) v4i_vs* g128p;
typedef __attribute__((address_space(3))) v4i_vs* l128p;
#else
#define USE_ASYNC_LDS 0
#endif

__device__ __forceinline__ unsigned short f32_to_bf16(float f) {
  unsigned int x = __float_as_uint(f);
  x += 0x7fffu + ((x >> 16) & 1u);          // round to nearest even
  return (unsigned short)(x >> 16);
}

// ---------------------------------------------------------------------------
// GEMM: C[M x Nm] = A[M x K](bf16) * W[Nm x K]^T(bf16) + bias, epilogue op.
// op: 0 = none, 1 = relu, 2 = softplus. Optional bf16 mirror store (Cbf).
// All K dims here are multiples of 16; all leading dims multiples of 8, so
// 16-short chunks are aligned and chunk-granular zero-padding is exact.
// ---------------------------------------------------------------------------
#define BLK_M 128
#define BLK_N 128
#define BLK_K 32
#define LDSS  40   // shorts per LDS row (80B): 16B-aligned rows, bank-spread

__global__ void __launch_bounds__(256)
gemm_bf16_kernel(const unsigned short* __restrict__ A, int lda,
                 const unsigned short* __restrict__ W, int ldw,
                 const float* __restrict__ bias,
                 float* __restrict__ C, int ldc,
                 unsigned short* __restrict__ Cbf,
                 int M, int K, int Nm, int op)
{
  __shared__ alignas(16) unsigned short sA[BLK_M * LDSS];
  __shared__ alignas(16) unsigned short sB[BLK_N * LDSS];

  const int tid  = threadIdx.x;
  const int lane = tid & 31;
  const int wave = tid >> 5;
  const int wm   = wave & 3;     // 4 waves along M (32 rows each)
  const int wn   = wave >> 2;    // 2 waves along N (64 cols each)
  const int bm   = blockIdx.y * BLK_M;
  const int bn   = blockIdx.x * BLK_N;

  const v8f vzero = {0.f,0.f,0.f,0.f,0.f,0.f,0.f,0.f};
  v8f acc[2][4];
  #pragma unroll
  for (int sm = 0; sm < 2; ++sm)
    #pragma unroll
    for (int sn = 0; sn < 4; ++sn) acc[sm][sn] = vzero;

  const int l16  = lane & 15;
  const int half = lane >> 4;          // lane group 0/1
  const int ka_u = half * 8;           // A frag: K 0-15 vs 16-31
  const int kb_u = half * 8;           // B frag: K 0-15 vs 16-31

  // Staging: one 16-short (32B) chunk per thread per tile.
  const int srow  = tid >> 1;          // 0..127
  const int shalf = tid & 1;           // which 16-wide K half

  for (int kt = 0; kt < K; kt += BLK_K) {
    const int gk = kt + shalf * 16;
    const bool va = ((bm + srow) < M)  && ((gk + 16) <= K);
    const bool vb = ((bn + srow) < Nm) && ((gk + 16) <= K);
    unsigned short* dap = &sA[srow * LDSS + shalf * 16];
    unsigned short* dbp = &sB[srow * LDSS + shalf * 16];

#if USE_ASYNC_LDS
    __syncthreads();   // previous tile fully consumed
    if (va) {
      const unsigned short* sa = A + (size_t)(bm + srow) * lda + gk;
      __builtin_amdgcn_global_load_async_to_lds_b128((g128p)sa,       (l128p)dap,       0, 0);
      __builtin_amdgcn_global_load_async_to_lds_b128((g128p)(sa + 8), (l128p)(dap + 8), 0, 0);
    } else {
      uint4 z = make_uint4(0u,0u,0u,0u);
      ((uint4*)dap)[0] = z; ((uint4*)dap)[1] = z;
    }
    if (vb) {
      const unsigned short* sb = W + (size_t)(bn + srow) * ldw + gk;
      __builtin_amdgcn_global_load_async_to_lds_b128((g128p)sb,       (l128p)dbp,       0, 0);
      __builtin_amdgcn_global_load_async_to_lds_b128((g128p)(sb + 8), (l128p)(dbp + 8), 0, 0);
    } else {
      uint4 z = make_uint4(0u,0u,0u,0u);
      ((uint4*)dbp)[0] = z; ((uint4*)dbp)[1] = z;
    }
    __builtin_amdgcn_s_wait_asynccnt(0);
    __syncthreads();
#else
    uint4 av0 = make_uint4(0u,0u,0u,0u), av1 = av0, bv0 = av0, bv1 = av0;
    if (va) {
      const uint4* p = (const uint4*)(A + (size_t)(bm + srow) * lda + gk);
      av0 = p[0]; av1 = p[1];
    }
    if (vb) {
      const uint4* p = (const uint4*)(W + (size_t)(bn + srow) * ldw + gk);
      bv0 = p[0]; bv1 = p[1];
    }
    __syncthreads();   // previous tile fully consumed
    ((uint4*)dap)[0] = av0; ((uint4*)dap)[1] = av1;
    ((uint4*)dbp)[0] = bv0; ((uint4*)dbp)[1] = bv1;
    __syncthreads();
#endif

    // Fragments: lane half h holds K 16h..16h+15 contiguously, same order for A and B.
    Frag16 a[2], b[4];
    #pragma unroll
    for (int sm = 0; sm < 2; ++sm) {
      const unsigned int* p = (const unsigned int*)&sA[(wm*32 + sm*16 + l16) * LDSS];
      #pragma unroll
      for (int j = 0; j < 4; ++j) {
        a[sm].u[j]     = p[ka_u + j];
        a[sm].u[4 + j] = p[ka_u + 4 + j];
      }
    }
    #pragma unroll
    for (int sn = 0; sn < 4; ++sn) {
      const unsigned int* q = (const unsigned int*)&sB[(wn*64 + sn*16 + l16) * LDSS];
      #pragma unroll
      for (int j = 0; j < 8; ++j) b[sn].u[j] = q[kb_u + j];
    }
    #pragma unroll
    for (int sm = 0; sm < 2; ++sm)
      #pragma unroll
      for (int sn = 0; sn < 4; ++sn)
        acc[sm][sn] = __builtin_amdgcn_wmma_f32_16x16x32_bf16(
            false, a[sm].v, false, b[sn].v, (short)0, acc[sm][sn], false, false);
  }

  // C/D layout: lane 0-15 -> N=lane, M=vgpr; lane 16-31 -> N=lane-16, M=8+vgpr.
  const int row0 = bm + wm * 32 + half * 8;
  const int col0 = bn + wn * 64 + l16;
  #pragma unroll
  for (int sm = 0; sm < 2; ++sm) {
    #pragma unroll
    for (int sn = 0; sn < 4; ++sn) {
      int cc = col0 + sn * 16;
      if (cc < Nm) {
        float bv = bias ? bias[cc] : 0.0f;
        #pragma unroll
        for (int v = 0; v < 8; ++v) {
          int rr = row0 + sm * 16 + v;
          if (rr < M) {
            float x = acc[sm][sn][v] + bv;
            if (op == 1)      x = fmaxf(x, 0.0f);
            else if (op == 2) x = (x > 20.0f) ? x : log1pf(__expf(x));
            C[(size_t)rr * ldc + cc] = x;
            if (Cbf) Cbf[(size_t)rr * ldc + cc] = f32_to_bf16(x);
          }
        }
      }
    }
  }
}

// ---------------------------------------------------------------------------
// Elementwise / helper kernels
// ---------------------------------------------------------------------------
__global__ void cvt_bf16_kernel(const float* __restrict__ s,
                                unsigned short* __restrict__ d, int n) {
  int i = blockIdx.x * 256 + threadIdx.x;
  if (i < n) d[i] = f32_to_bf16(s[i]);
}

__global__ void flip_kernel(const float* __restrict__ x,
                            float* __restrict__ xf,
                            unsigned short* __restrict__ xfb) {
  int i = blockIdx.x * 256 + threadIdx.x;
  if (i >= NTOK * DMODEL) return;
  int c = i % DMODEL;
  int row = i / DMODEL;
  int t = row % LSEQ, b = row / LSEQ;
  float v = x[((size_t)(b * LSEQ + (LSEQ - 1 - t))) * DMODEL + c];
  xf[i] = v;
  xfb[i] = f32_to_bf16(v);
}

// Depthwise causal conv (D_CONV=4) + SiLU over the xi half of xz.
__global__ void conv_silu_kernel(const float* __restrict__ xz,
                                 const float* __restrict__ w,
                                 const float* __restrict__ bias,
                                 float* __restrict__ xc,
                                 unsigned short* __restrict__ xcb) {
  int i = blockIdx.x * 256 + threadIdx.x;
  if (i >= NTOK * DINNER) return;
  int c   = i % DINNER;
  int row = i / DINNER;
  int t = row % LSEQ, b = row / LSEQ;
  float s = bias[c];
  #pragma unroll
  for (int j = 0; j < 4; ++j) {
    int tt = t - 3 + j;
    if (tt >= 0) s += w[c * 4 + j] * xz[((size_t)(b * LSEQ + tt)) * (2 * DINNER) + c];
  }
  float v = s / (1.f + __expf(-s));        // SiLU
  xc[i]  = v;
  xcb[i] = f32_to_bf16(v);
}

// Selective scan: one thread per (batch, channel); 16 states in registers.
// B/C (shared by all channels of a token) staged in LDS per 64-step chunk.
__global__ void __launch_bounds__(256)
scan_kernel(const float* __restrict__ u, const float* __restrict__ delta,
            const float* __restrict__ dbc, const float* __restrict__ A_log,
            const float* __restrict__ Dp, float* __restrict__ y)
{
  __shared__ float sBC[64 * 32];           // 64 timesteps x (B[16] | C[16])
  const int tid = threadIdx.x;
  const int gid = blockIdx.x * 256 + tid;  // all threads in block share b
  const int c = gid % DINNER;
  const int b = gid / DINNER;

  float Av[DSTATE];
  #pragma unroll
  for (int s = 0; s < DSTATE; ++s) Av[s] = -__expf(A_log[c * DSTATE + s]);
  const float Dc = Dp[c];

  float h[DSTATE];
  #pragma unroll
  for (int s = 0; s < DSTATE; ++s) h[s] = 0.f;

  for (int t0 = 0; t0 < LSEQ; t0 += 64) {
    __syncthreads();
    for (int i = tid; i < 64 * 32; i += 256) {
      int tt = i >> 5, j = i & 31;
      sBC[i] = dbc[((size_t)(b * LSEQ + t0 + tt)) * DBCW + DTRANK + j];
    }
    __syncthreads();
    for (int tt = 0; tt < 64; ++tt) {
      size_t idx = ((size_t)(b * LSEQ + t0 + tt)) * DINNER + c;
      float dlt = delta[idx];
      float uu  = u[idx];
      float du  = dlt * uu;
      const float* bc = &sBC[tt * 32];     // broadcast reads
      float acc = 0.f;
      #pragma unroll
      for (int s = 0; s < DSTATE; ++s) {
        h[s] = __expf(dlt * Av[s]) * h[s] + du * bc[s];
        acc += h[s] * bc[16 + s];
      }
      y[idx] = acc + uu * Dc;
    }
  }
}

// out = y*silu(z) + xc*(1-silu(z)), emitted directly as bf16 for out_proj.
__global__ void gate_kernel(const float* __restrict__ y,
                            const float* __restrict__ xz,
                            const float* __restrict__ xc,
                            unsigned short* __restrict__ outb) {
  int i = blockIdx.x * 256 + threadIdx.x;
  if (i >= NTOK * DINNER) return;
  int row = i / DINNER, c = i % DINNER;
  float z  = xz[(size_t)row * (2 * DINNER) + DINNER + c];
  float zg = z / (1.f + __expf(-z));
  float v  = y[i] * zg + xc[i] * (1.f - zg);
  outb[i] = f32_to_bf16(v);
}

// out = LayerNorm(src[flip?] + res) * g + b ; one block per 768-wide row.
__global__ void __launch_bounds__(256)
ln_add_kernel(const float* __restrict__ src, const float* __restrict__ res,
              const float* __restrict__ g, const float* __restrict__ bta,
              float* __restrict__ out, unsigned short* __restrict__ outb,
              int flip)
{
  __shared__ float s1[256], s2[256];
  const int tid = threadIdx.x;
  const int row = blockIdx.x;              // 0..NTOK-1
  const int t = row % LSEQ, b = row / LSEQ;
  const int srow = flip ? (b * LSEQ + (LSEQ - 1 - t)) : row;
  const float* ps = src + (size_t)srow * DMODEL;
  const float* pr = res + (size_t)row * DMODEL;

  float v[3], lsum = 0.f, lsq = 0.f;
  #pragma unroll
  for (int j = 0; j < 3; ++j) {
    int c = tid + j * 256;
    float x = ps[c] + pr[c];
    v[j] = x; lsum += x; lsq += x * x;
  }
  s1[tid] = lsum; s2[tid] = lsq;
  __syncthreads();
  for (int off = 128; off > 0; off >>= 1) {
    if (tid < off) { s1[tid] += s1[tid + off]; s2[tid] += s2[tid + off]; }
    __syncthreads();
  }
  float mu   = s1[0] * (1.f / DMODEL);
  float var  = s2[0] * (1.f / DMODEL) - mu * mu;
  float rstd = rsqrtf(var + 1e-5f);
  #pragma unroll
  for (int j = 0; j < 3; ++j) {
    int c = tid + j * 256;
    float x = (v[j] - mu) * rstd * g[c] + bta[c];
    out[(size_t)row * DMODEL + c] = x;
    if (outb) outb[(size_t)row * DMODEL + c] = f32_to_bf16(x);
  }
}

__global__ void add_h_kernel(const float* __restrict__ a, const float* __restrict__ b,
                             float* __restrict__ h, unsigned short* __restrict__ hb,
                             int n) {
  int i = blockIdx.x * 256 + threadIdx.x;
  if (i < n) { float v = a[i] + b[i]; h[i] = v; hb[i] = f32_to_bf16(v); }
}

// ---------------------------------------------------------------------------
// Host orchestration
// ---------------------------------------------------------------------------
extern "C" void kernel_launch(void* const* d_in, const int* in_sizes, int n_in,
                              void* d_out, int out_size, void* d_ws, size_t ws_size,
                              hipStream_t stream) {
  (void)in_sizes; (void)n_in; (void)out_size; (void)ws_size;

  const float* x_in = (const float*)d_in[0];
  // fwd base 1, bwd base 12: +0 in_proj_w, +1 in_proj_b, +2 conv_w, +3 conv_b,
  // +4 x_proj_w, +5 dt_proj_w, +6 dt_proj_b, +7 A_log, +8 D, +9 out_proj_w, +10 out_proj_b
  const int FB = 1, BB = 12;
  const float* g1 = (const float*)d_in[23]; const float* n_b1 = (const float*)d_in[24];
  const float* g2 = (const float*)d_in[25]; const float* n_b2 = (const float*)d_in[26];
  const float* g3 = (const float*)d_in[27]; const float* n_b3 = (const float*)d_in[28];
  const float* ffn_w1 = (const float*)d_in[29]; const float* ffn_b1 = (const float*)d_in[30];
  const float* ffn_w2 = (const float*)d_in[31]; const float* ffn_b2 = (const float*)d_in[32];

  char* base = (char*)d_ws;
  size_t cur = 0;
  auto alloc = [&](size_t bytes) -> char* {
    char* p = base + cur;
    cur += (bytes + 255) & ~(size_t)255;
    return p;
  };
  unsigned short *wbf_in[2], *wbf_xp[2], *wbf_dt[2], *wbf_out[2];
  for (int d = 0; d < 2; ++d) {
    wbf_in[d]  = (unsigned short*)alloc((size_t)2 * DINNER * DMODEL * 2);
    wbf_xp[d]  = (unsigned short*)alloc((size_t)DBCW * DINNER * 2);
    wbf_dt[d]  = (unsigned short*)alloc((size_t)DINNER * DTRANK * 2);
    wbf_out[d] = (unsigned short*)alloc((size_t)DMODEL * DINNER * 2);
  }
  unsigned short* wbf_ff1 = (unsigned short*)alloc((size_t)DFF * DMODEL * 2);
  unsigned short* wbf_ff2 = (unsigned short*)alloc((size_t)DMODEL * DFF * 2);
  unsigned short* x_bf    = (unsigned short*)alloc((size_t)NTOK * DMODEL * 2);
  float* xflip            = (float*)alloc((size_t)NTOK * DMODEL * 4);
  unsigned short* xflip_bf= (unsigned short*)alloc((size_t)NTOK * DMODEL * 2);
  float* xz               = (float*)alloc((size_t)NTOK * 2 * DINNER * 4); // also ff1 f32
  float* xc               = (float*)alloc((size_t)NTOK * DINNER * 4);
  unsigned short* xc_bf   = (unsigned short*)alloc((size_t)NTOK * DINNER * 2);
  float* dbc              = (float*)alloc((size_t)NTOK * DBCW * 4);
  unsigned short* dbc_bf  = (unsigned short*)alloc((size_t)NTOK * DBCW * 2);
  float* delta            = (float*)alloc((size_t)NTOK * DINNER * 4);
  float* ybuf             = (float*)alloc((size_t)NTOK * DINNER * 4);
  unsigned short* gated_bf= (unsigned short*)alloc((size_t)NTOK * DINNER * 2);
  float* mamba_f          = (float*)alloc((size_t)NTOK * DMODEL * 4);
  float* mamba_b          = (float*)alloc((size_t)NTOK * DMODEL * 4);
  float* x_f              = (float*)alloc((size_t)NTOK * DMODEL * 4);
  float* x_b              = (float*)alloc((size_t)NTOK * DMODEL * 4);
  float* hbuf             = (float*)alloc((size_t)NTOK * DMODEL * 4);
  unsigned short* h_bf    = (unsigned short*)alloc((size_t)NTOK * DMODEL * 2);
  unsigned short* ff1_bf  = (unsigned short*)alloc((size_t)NTOK * DFF * 2);
  float* ff2              = (float*)alloc((size_t)NTOK * DMODEL * 4);

  auto cvt = [&](const float* s, unsigned short* d, size_t n) {
    cvt_bf16_kernel<<<(unsigned)((n + 255) / 256), 256, 0, stream>>>(s, d, (int)n);
  };
  auto gemm = [&](const unsigned short* A, int lda, const unsigned short* W, int ldw,
                  const float* bias, float* C, int ldc, unsigned short* Cbf,
                  int M, int K, int Nm, int op) {
    dim3 g((Nm + BLK_N - 1) / BLK_N, (M + BLK_M - 1) / BLK_M);
    gemm_bf16_kernel<<<g, 256, 0, stream>>>(A, lda, W, ldw, bias, C, ldc, Cbf, M, K, Nm, op);
  };

  for (int d = 0; d < 2; ++d) {
    int bb = d ? BB : FB;
    cvt((const float*)d_in[bb + 0], wbf_in[d],  (size_t)2 * DINNER * DMODEL);
    cvt((const float*)d_in[bb + 4], wbf_xp[d],  (size_t)DBCW * DINNER);
    cvt((const float*)d_in[bb + 5], wbf_dt[d],  (size_t)DINNER * DTRANK);
    cvt((const float*)d_in[bb + 9], wbf_out[d], (size_t)DMODEL * DINNER);
  }
  cvt(ffn_w1, wbf_ff1, (size_t)DFF * DMODEL);
  cvt(ffn_w2, wbf_ff2, (size_t)DMODEL * DFF);
  cvt(x_in, x_bf, (size_t)NTOK * DMODEL);
  flip_kernel<<<(NTOK * DMODEL) / 256, 256, 0, stream>>>(x_in, xflip, xflip_bf);

  auto mamba = [&](const unsigned short* xin_bf, int bb, int d, float* mout) {
    const float* b_in   = (const float*)d_in[bb + 1];
    const float* conv_w = (const float*)d_in[bb + 2];
    const float* conv_b = (const float*)d_in[bb + 3];
    const float* b_dt   = (const float*)d_in[bb + 6];
    const float* A_log  = (const float*)d_in[bb + 7];
    const float* Dp     = (const float*)d_in[bb + 8];
    const float* b_out  = (const float*)d_in[bb + 10];
    gemm(xin_bf, DMODEL, wbf_in[d], DMODEL, b_in, xz, 2 * DINNER, nullptr,
         NTOK, DMODEL, 2 * DINNER, 0);
    conv_silu_kernel<<<(NTOK * DINNER) / 256, 256, 0, stream>>>(xz, conv_w, conv_b, xc, xc_bf);
    gemm(xc_bf, DINNER, wbf_xp[d], DINNER, nullptr, dbc, DBCW, dbc_bf,
         NTOK, DINNER, DBCW, 0);
    gemm(dbc_bf, DBCW, wbf_dt[d], DTRANK, b_dt, delta, DINNER, nullptr,
         NTOK, DTRANK, DINNER, 2);
    scan_kernel<<<(NBATCH * DINNER) / 256, 256, 0, stream>>>(xc, delta, dbc, A_log, Dp, ybuf);
    gate_kernel<<<(NTOK * DINNER) / 256, 256, 0, stream>>>(ybuf, xz, xc, gated_bf);
    gemm(gated_bf, DINNER, wbf_out[d], DINNER, b_out, mout, DMODEL, nullptr,
         NTOK, DINNER, DMODEL, 0);
  };

  mamba(x_bf,     FB, 0, mamba_f);
  mamba(xflip_bf, BB, 1, mamba_b);

  ln_add_kernel<<<NTOK, 256, 0, stream>>>(mamba_f, x_in, g1, n_b1, x_f, nullptr, 0);
  ln_add_kernel<<<NTOK, 256, 0, stream>>>(mamba_b, x_in, g2, n_b2, x_b, nullptr, 1);
  add_h_kernel<<<(NTOK * DMODEL) / 256, 256, 0, stream>>>(x_f, x_b, hbuf, h_bf, NTOK * DMODEL);

  gemm(h_bf, DMODEL, wbf_ff1, DMODEL, ffn_b1, xz, DFF, ff1_bf, NTOK, DMODEL, DFF, 1);
  gemm(ff1_bf, DFF, wbf_ff2, DFF, ffn_b2, ff2, DMODEL, nullptr, NTOK, DFF, DMODEL, 0);

  ln_add_kernel<<<NTOK, 256, 0, stream>>>(ff2, hbuf, g3, n_b3, (float*)d_out, nullptr, 0);
}
